// VisionEncoder_4277787427229
// MI455X (gfx1250) — matrix operand order repaired and checked by reading, not compiled, not run on hardware
//
#include <hip/hip_runtime.h>
#include <hip/hip_bf16.h>

// ---------------------------------------------------------------------------
// VisionEncoder fused pipeline for gfx1250 (MI455X, wave32, WMMA + TDM)
//   feat = concat(c3, bilinear_up2x(c4))            (8,1536,64,64)
//   att  = relu(BN(1x1conv(feat)))                  (8,512,64,64)
//   out  = (boxes, roi_adaptive_pool5x5(att,boxes)) (8,10,512,5,5)
// GEMM: M=32768 (spatial), N=512 (out ch), K=1536 via V_WMMA_F32_16X16X32_F16.
// B (weights) staged to LDS per 128-K chunk with the Tensor Data Mover,
// double buffered; LDS rows padded by the TDM pad feature for conflict-free
// ds_load_b128 fragment reads; B fragments software-pipelined one WMMA ahead.
// ---------------------------------------------------------------------------

typedef __attribute__((ext_vector_type(16))) _Float16 v16h;
typedef __attribute__((ext_vector_type(8)))  _Float16 v8h;
typedef __attribute__((ext_vector_type(8)))  float    v8f;
typedef unsigned int u32x4 __attribute__((ext_vector_type(4)));
typedef int          i32x8 __attribute__((ext_vector_type(8)));
typedef int          i32x4 __attribute__((ext_vector_type(4)));

#define GEMM_M 32768   // 8*64*64
#define GEMM_N 512
#define GEMM_K 1536
#define IMG_B 8
#define IMG_HW 64
#define POOL 5
#define NOBJ 10
#define BN_EPS 1e-5f

#define CHUNK_K 128                      // K per LDS staging chunk
#define NCHUNK  (GEMM_K / CHUNK_K)       // 12
#define BROW    136                      // 128 halfs + 8 pad halfs (16B) per row

#if defined(__has_builtin)
# if __has_builtin(__builtin_amdgcn_tensor_load_to_lds) && \
     __has_builtin(__builtin_amdgcn_s_wait_tensorcnt)
#  define HAVE_TDM 1
# endif
#endif
#ifndef HAVE_TDM
# define HAVE_TDM 0
#endif

// ---------------------------------------------------------------------------
// Kernel 1: build fp16 A matrix  A[m][k], m = b*4096 + y*64 + x, k = channel.
//   k <  512 : c3[b][k][y][x]
//   k >= 512 : bilinear 2x upsample of c4[b][k-512] at (y,x)
// LDS-tiled so both the NCHW reads and the [m][k] writes are coalesced.
// grid = (M/32, K/64), block = 256.
// ---------------------------------------------------------------------------
__global__ void pack_feat_kernel(const float* __restrict__ c3,
                                 const float* __restrict__ c4,
                                 _Float16* __restrict__ A) {
  __shared__ _Float16 tile[64][33];           // [c_local][m_local], padded
  const int m0 = blockIdx.x * 32;
  const int c0 = blockIdx.y * 64;
  const int b  = m0 >> 12;                    // m0 / 4096
  const int y  = (m0 & 4095) >> 6;
  const int x0 = m0 & 63;                     // 0 or 32 (tiles never cross a row)

  const int mi   = threadIdx.x & 31;
  const int cseg = threadIdx.x >> 5;          // 0..7
  const int x    = x0 + mi;

  const float fy = 0.5f * (float)y - 0.25f;
  const int   jy = (int)floorf(fy);
  const float ty = fy - (float)jy;
  const int   y0c = jy < 0 ? 0 : jy;
  const int   y1c = (jy + 1) > 31 ? 31 : (jy + 1);
  const float fx = 0.5f * (float)x - 0.25f;
  const int   jx = (int)floorf(fx);
  const float tx = fx - (float)jx;
  const int   x0c = jx < 0 ? 0 : jx;
  const int   x1c = (jx + 1) > 31 ? 31 : (jx + 1);

  #pragma unroll
  for (int p = 0; p < 8; ++p) {
    const int ci = cseg + p * 8;              // 0..63
    const int c  = c0 + ci;
    float v;
    if (c < 512) {
      v = c3[((size_t)(b * 512 + c) << 12) + (y << 6) + x];
    } else {
      const float* base = c4 + ((size_t)(b * 1024 + (c - 512)) << 10);
      const float v00 = base[(y0c << 5) + x0c];
      const float v01 = base[(y0c << 5) + x1c];
      const float v10 = base[(y1c << 5) + x0c];
      const float v11 = base[(y1c << 5) + x1c];
      v = (1.f - ty) * ((1.f - tx) * v00 + tx * v01) +
          ty         * ((1.f - tx) * v10 + tx * v11);
    }
    tile[ci][mi] = (_Float16)v;
  }
  __syncthreads();

  const int ci2 = threadIdx.x & 63;
  const int ms  = threadIdx.x >> 6;           // 0..3
  #pragma unroll
  for (int p = 0; p < 8; ++p) {
    const int mi2 = ms + p * 4;               // 0..31
    A[(size_t)(m0 + mi2) * GEMM_K + c0 + ci2] = tile[ci2][mi2];
  }
}

// ---------------------------------------------------------------------------
// Kernel 2: conv_w fp32 -> fp16, row-major [o][k] (== B-matrix columns)
// ---------------------------------------------------------------------------
__global__ void pack_w_kernel(const float* __restrict__ w,
                              _Float16* __restrict__ wh, int n) {
  int i = blockIdx.x * 256 + threadIdx.x;
  if (i < n) wh[i] = (_Float16)w[i];
}

// ---------------------------------------------------------------------------
// TDM helper: issue a tensor_load_to_lds of a 64-row x 128-half B chunk.
// D# per CDNA5 ISA 8.3/8.4:
//   group0: count=1, lds_addr, global_addr, type=2
//   group1: data_size=2B, pad_enable (64 DWORDs interval, 4 DWORDs pad),
//           tensor_dim0=128, tensor_dim1=64, tile_dim0=128, tile_dim1=64,
//           tensor_dim0_stride=1536
// Padded LDS row stride = 64+4 = 68 DWORDs = BROW halfs -> conflict-free
// ds_load_b128 of B fragments.
// 6-arg builtin on this toolchain:
//   (u32x4 g0, i32x8 g1, i32x4 g2, i32x4 g3, i32x8 extra, i32 cpol)
// ---------------------------------------------------------------------------
#if HAVE_TDM
__device__ __forceinline__ void tdm_load_btile(const _Float16* gsrc,
                                               const void* ldsdst) {
  const unsigned long long ga = (unsigned long long)(uintptr_t)gsrc;
  const unsigned int lds_off = (unsigned int)(uintptr_t)ldsdst; // LDS byte addr

  u32x4 g0;
  g0[0] = 1u;                                       // count=1, user descriptor
  g0[1] = lds_off;                                  // lds_addr
  g0[2] = (unsigned int)ga;                         // global_addr[31:0]
  g0[3] = (unsigned int)((ga >> 32) & 0x01FFFFFFu)  // global_addr[56:32]
        | (2u << 30);                               // type=2 ("image")

  i32x8 g1;
  g1[0] = (int)((1u << 16)      // data_size = 1 -> 2 bytes
              | (1u << 20)      // pad_enable
              | (5u << 22)      // pad_interval: 2^(5+1)=64 DWORDs (one row)
              | (3u << 25));    // pad_amount: 3+1 = 4 DWORDs (16B)
  g1[1] = (int)(128u << 16);    // tensor_dim0[15:0]=128 (bits 63:48)
  g1[2] = (int)(64u << 16);     // tensor_dim1[15:0]=64  (bits 95:80)
  g1[3] = (int)(128u << 16);    // tile_dim0=128         (bits 127:112)
  g1[4] = 64;                   // tile_dim1=64          (bits 143:128)
  g1[5] = GEMM_K;               // tensor_dim0_stride=1536 (bits 191:160)
  g1[6] = 0;
  g1[7] = 0;

  i32x4 gz4 = {0, 0, 0, 0};     // groups 2/3 unused (2D tile)
  i32x8 gz8 = {0, 0, 0, 0, 0, 0, 0, 0};
  __builtin_amdgcn_tensor_load_to_lds(g0, g1, gz4, gz4, gz8, 0);
}
#endif

// ---------------------------------------------------------------------------
// Kernel 3: WMMA GEMM + BN + ReLU.  att[m][o] fp16 (m-major for the pooler).
// Wave tile: 16M x 64N (4 accumulators reuse one A fragment).
// Block: 8 waves -> 128M x 64N.  grid = (M/128, N/64).
// B staged in LDS (TDM double-buffered), A register-prefetched per chunk,
// B fragments software-pipelined one WMMA ahead (waits become dscnt<=2).
// ---------------------------------------------------------------------------
union AFrag { v16h v; v8h h[2]; };

__global__ void __launch_bounds__(256)
gemm_bn_relu_kernel(const _Float16* __restrict__ A,
                    const _Float16* __restrict__ Wh,
                    const float* __restrict__ conv_b,
                    const float* __restrict__ bn_gamma,
                    const float* __restrict__ bn_beta,
                    const float* __restrict__ bn_mean,
                    const float* __restrict__ bn_var,
                    _Float16* __restrict__ att) {
  __shared__ _Float16 bbuf[2][64 * BROW];   // 2 x 17408 B

  const int wave = threadIdx.x >> 5;
  const int lane = threadIdx.x & 31;
  const int r    = lane & 15;
  const int hseg = lane >> 4;

  const int m0 = blockIdx.x * 128 + wave * 16;
  const int n0 = blockIdx.y * 64;

  // A: lane holds row M=r; hseg=0 -> K {kb+0..7, kb+16..23}, hseg=1 -> {kb+8..15, kb+24..31}
  const _Float16* __restrict__ Arow = A + (size_t)(m0 + r) * GEMM_K + hseg * 8;
  const _Float16* __restrict__ Bsrc = Wh + (size_t)n0 * GEMM_K;

#if HAVE_TDM
  if (threadIdx.x < 32) {                       // one TDM op per chunk (wave 0)
    tdm_load_btile(Bsrc, &bbuf[0][0]);
    __builtin_amdgcn_s_wait_tensorcnt(0);
  }
  __syncthreads();
#endif

  v8f acc0 = {}, acc1 = {}, acc2 = {}, acc3 = {};

  for (int kc = 0; kc < NCHUNK; ++kc) {
    const int cur = kc & 1;
#if HAVE_TDM
    if (kc + 1 < NCHUNK && threadIdx.x < 32)    // prefetch next chunk
      tdm_load_btile(Bsrc + (size_t)(kc + 1) * CHUNK_K, &bbuf[cur ^ 1][0]);
#else
    // cooperative fallback staging (no TDM builtin on this toolchain)
    if (kc > 0) __syncthreads();                // protect buffer reuse
    {
      const int row = threadIdx.x >> 2;         // 0..63
      const int seg = threadIdx.x & 3;          // 0..3 (32 halfs each)
      const _Float16* gs = Bsrc + (size_t)row * GEMM_K + kc * CHUNK_K + seg * 32;
      _Float16* ds = &bbuf[cur][row * BROW + seg * 32];
      #pragma unroll
      for (int q = 0; q < 4; ++q)
        *(v8h*)(ds + q * 8) = *(const v8h*)(gs + q * 8);
    }
    __syncthreads();
#endif

    // register-prefetch all 4 A fragments of this chunk (8 x b128, one clause)
    const _Float16* Ac = Arow + kc * CHUNK_K;
    v8h a_lo[4], a_hi[4];
    #pragma unroll
    for (int ks = 0; ks < 4; ++ks) {
      a_lo[ks] = *(const v8h*)(Ac + ks * 32);
      a_hi[ks] = *(const v8h*)(Ac + ks * 32 + 16);
    }

    // Software-pipelined 16-step WMMA sequence: idx -> (ks = idx>>2, t = idx&3).
    // B fragment for step idx+1 is loaded before WMMA idx issues.
    const _Float16* __restrict__ Bbase =
        &bbuf[cur][(size_t)r * BROW + hseg * 8];
    AFrag b_cur, b_nxt;
    b_cur.h[0] = *(const v8h*)(Bbase);
    b_cur.h[1] = *(const v8h*)(Bbase + 16);
    b_nxt = b_cur;

    #pragma unroll
    for (int idx = 0; idx < 16; ++idx) {
      const int ks = idx >> 2;
      const int t  = idx & 3;
      if (idx < 15) {
        const int ni  = idx + 1;
        const int nks = ni >> 2;
        const int nt  = ni & 3;
        const _Float16* bq = Bbase + nks * 32 + nt * (16 * BROW);
        b_nxt.h[0] = *(const v8h*)(bq);
        b_nxt.h[1] = *(const v8h*)(bq + 16);
      }
      AFrag a; a.h[0] = a_lo[ks]; a.h[1] = a_hi[ks];
      v8f* accp = (t == 0) ? &acc0 : (t == 1) ? &acc1 : (t == 2) ? &acc2 : &acc3;
      *accp = __builtin_amdgcn_wmma_f32_16x16x32_f16(false, a.v, false, b_cur.v,
                                                     (short)0, *accp,
                                                     false, false);
      b_cur = b_nxt;
    }

#if HAVE_TDM
    if (threadIdx.x < 32 && kc + 1 < NCHUNK)
      __builtin_amdgcn_s_wait_tensorcnt(0);     // next chunk landed
    __syncthreads();                            // publish to all waves
#endif
  }

  // Epilogue: y = s*(acc + conv_b - mean) + beta, s = gamma*rsqrt(var+eps); relu.
  // C/D layout: lane (hseg,r): element j -> M = m0 + hseg*8 + j, N = och.
  v8f accs[4] = {acc0, acc1, acc2, acc3};
  #pragma unroll
  for (int t = 0; t < 4; ++t) {
    const int och = n0 + 16 * t + r;
    const float s  = bn_gamma[och] * rsqrtf(bn_var[och] + BN_EPS);
    const float bs = bn_beta[och] + (conv_b[och] - bn_mean[och]) * s;
    #pragma unroll
    for (int j = 0; j < 8; ++j) {
      const int m = m0 + hseg * 8 + j;
      float v = accs[t][j] * s + bs;
      v = v > 0.f ? v : 0.f;
      att[(size_t)m * GEMM_N + och] = (_Float16)v;
    }
  }
}

// ---------------------------------------------------------------------------
// Kernel 4: ROI adaptive 5x5 average pool.
// One block per (b, obj, py, px) bin; threads stride channels (coalesced att).
// ---------------------------------------------------------------------------
__global__ void roi_pool_kernel(const _Float16* __restrict__ att,
                                const float* __restrict__ boxes,
                                float* __restrict__ out) {
  const int bid = blockIdx.x;          // b*250 + obj*25 + py*5 + px
  const int px  = bid % 5;
  const int py  = (bid / 5) % 5;
  const int obj = (bid / 25) % NOBJ;
  const int b   = bid / (25 * NOBJ);

  const float* box = boxes + (size_t)(b * NOBJ + obj) * 4;
  const float scale = (float)IMG_HW / 512.0f;   // H / ORIG_H
  int x1 = (int)floorf(box[0] * scale);
  int y1 = (int)floorf(box[1] * scale);
  int x2 = (int)ceilf(box[2] * scale);
  int y2 = (int)ceilf(box[3] * scale);
  x1 = x1 < 0 ? 0 : (x1 > 63 ? 63 : x1);
  y1 = y1 < 0 ? 0 : (y1 > 63 ? 63 : y1);
  x2 = x2 < 0 ? 0 : (x2 > 63 ? 63 : x2);
  y2 = y2 < 0 ? 0 : (y2 > 63 ? 63 : y2);
  if (x2 <= x1) x2 = (x1 + 1) < 64 ? (x1 + 1) : 64;
  if (y2 <= y1) y2 = (y1 + 1) < 64 ? (y1 + 1) : 64;

  const int Ly = y2 - y1, Lx = x2 - x1;
  const int rs = y1 + (py * Ly) / POOL;
  const int re = y1 + ((py + 1) * Ly + POOL - 1) / POOL;
  const int cs = x1 + (px * Lx) / POOL;
  const int ce = x1 + ((px + 1) * Lx + POOL - 1) / POOL;
  const float inv = 1.0f / (float)((re - rs) * (ce - cs));

  for (int o = threadIdx.x; o < GEMM_N; o += 256) {
    float s = 0.f;
    for (int yy = rs; yy < re; ++yy) {
      const _Float16* row = att + ((size_t)((b << 12) + (yy << 6)) * GEMM_N) + o;
      for (int xx = cs; xx < ce; ++xx)
        s += (float)row[(size_t)xx * GEMM_N];
    }
    out[320 + ((((size_t)(b * NOBJ + obj) * GEMM_N + o) * POOL + py) * POOL + px)] =
        s * inv;
  }
}

__global__ void copy_boxes_kernel(const float* __restrict__ boxes,
                                  float* __restrict__ out) {
  int i = blockIdx.x * 64 + threadIdx.x;
  if (i < IMG_B * NOBJ * 4) out[i] = boxes[i];
}

// ---------------------------------------------------------------------------
extern "C" void kernel_launch(void* const* d_in, const int* in_sizes, int n_in,
                              void* d_out, int out_size, void* d_ws, size_t ws_size,
                              hipStream_t stream) {
  const float* c3       = (const float*)d_in[0];
  const float* c4       = (const float*)d_in[1];
  const float* boxes    = (const float*)d_in[2];
  const float* conv_w   = (const float*)d_in[3];
  const float* conv_b   = (const float*)d_in[4];
  const float* bn_gamma = (const float*)d_in[5];
  const float* bn_beta  = (const float*)d_in[6];
  const float* bn_mean  = (const float*)d_in[7];
  const float* bn_var   = (const float*)d_in[8];
  float* out = (float*)d_out;

  // Workspace layout (fp16): A[M][K] | Wh[N][K] | att[M][N]  (~136 MB)
  _Float16* A   = (_Float16*)d_ws;
  _Float16* Wh  = A  + (size_t)GEMM_M * GEMM_K;
  _Float16* att = Wh + (size_t)GEMM_N * GEMM_K;

  {
    dim3 grid(GEMM_M / 32, GEMM_K / 64);
    pack_feat_kernel<<<grid, 256, 0, stream>>>(c3, c4, A);
  }
  {
    int n = GEMM_N * GEMM_K;
    pack_w_kernel<<<(n + 255) / 256, 256, 0, stream>>>(conv_w, Wh, n);
  }
  {
    dim3 grid(GEMM_M / 128, GEMM_N / 64);
    gemm_bn_relu_kernel<<<grid, 256, 0, stream>>>(A, Wh, conv_b, bn_gamma,
                                                  bn_beta, bn_mean, bn_var, att);
  }
  copy_boxes_kernel<<<(IMG_B * NOBJ * 4 + 63) / 64, 64, 0, stream>>>(boxes, out);
  roi_pool_kernel<<<IMG_B * NOBJ * POOL * POOL, 256, 0, stream>>>(att, boxes, out);
}
